// GATCoxModel_90958817395482
// MI455X (gfx1250) — matrix operand order repaired and checked by reading, not compiled, not run on hardware
//
#include <hip/hip_runtime.h>
#include <hip/hip_bf16.h>
#include <math.h>

typedef __attribute__((ext_vector_type(16))) _Float16 v16h;
typedef __attribute__((ext_vector_type(8)))  _Float16 v8h;
typedef __attribute__((ext_vector_type(8)))  float    v8f;
typedef int v4i __attribute__((vector_size(16)));

#define NEG_SLOPE 0.2f

// Detect gfx1250 async global->LDS builtins (guarded; fallback = vector copy)
#if defined(__has_builtin)
#  if __has_builtin(__builtin_amdgcn_global_load_async_to_lds_b128)
#    define USE_ASYNC_LDS 1
#  endif
#endif
#ifndef USE_ASYNC_LDS
#  define USE_ASYNC_LDS 0
#endif

// ---------------------------------------------------------------------------
// WMMA fragment helpers (CDNA5 wave32 layouts per ISA 7.12.2)
// ---------------------------------------------------------------------------

// A fragment from f16 row-major [.., lda]: rows row0..+15, cols k0..+31.
// Per documented layout, lane l: row = l&15, K-halfs [koff, koff+8) and
// [16+koff, 16+koff+8) with koff = (l>=16)?8:0  ->  two contiguous 16B loads.
__device__ __forceinline__ v16h load_a_frag_h(const _Float16* __restrict__ A, int lda,
                                              int row0, int k0, int lane, int nrows) {
  int m = lane & 15;
  int koff = (lane & 16) ? 8 : 0;
  int row = row0 + m; if (row >= nrows) row = nrows - 1;
  const _Float16* rp = A + (size_t)row * lda + k0 + koff;
  v8h lo = *(const v8h*)rp;          // K = k0+koff .. +7
  v8h hi = *(const v8h*)(rp + 16);   // K = k0+16+koff .. +7
  v16h a;
#pragma unroll
  for (int i = 0; i < 8; ++i) { a[i] = lo[i]; a[8 + i] = hi[i]; }
  return a;
}

// C/D fragment store: lane holds N=lane&15, VGPR r -> M = r + 8*(lane>=16)
__device__ __forceinline__ void store_c_frag(float* __restrict__ C, int ldc,
                                             int row0, int n0, int lane, v8f acc,
                                             int nrows) {
  int n = lane & 15;
  int mb = (lane & 16) ? 8 : 0;
#pragma unroll
  for (int r = 0; r < 8; ++r) {
    int row = row0 + mb + r;
    if (row < nrows) C[(size_t)row * ldc + n0 + n] = acc[r];
  }
}

// Stage `nbytes` of pre-packed weights from global into LDS.
__device__ __forceinline__ void stage_lds(const _Float16* __restrict__ gp,
                                          char* lds, int nbytes, int tcount) {
#if USE_ASYNC_LDS
  {
    int chunks = (nbytes >> 4) / tcount;             // 16B chunks per thread
    __attribute__((address_space(1))) v4i* g =
        (__attribute__((address_space(1))) v4i*)(uintptr_t)gp + (size_t)threadIdx.x * chunks;
    __attribute__((address_space(3))) v4i* l =
        (__attribute__((address_space(3))) v4i*)(uintptr_t)lds + (size_t)threadIdx.x * chunks;
    for (int i = 0; i < chunks; ++i)
      __builtin_amdgcn_global_load_async_to_lds_b128(g + i, l + i, 0, 0);
    asm volatile("s_wait_asynccnt 0" ::: "memory");
  }
#else
  for (int i = threadIdx.x; i < nbytes / 16; i += tcount)
    ((uint4*)lds)[i] = ((const uint4*)gp)[i];
#endif
  __syncthreads();
}

// ---------------------------------------------------------------------------
// Weight packing: W (f32 row-major [K, ldb]) -> f16 fragments.
// P[((nt*KS + ks)*32 + lane)*16 + e] = W[(ks*32 + kb + e + (e&8))*ldb + nt*16 + n]
//   n = lane&15, kb = (lane&16)?8:0
// ---------------------------------------------------------------------------
__global__ void k_pack_w(const float* __restrict__ W, _Float16* __restrict__ P,
                         int K, int ldb) {
  int ntiles = ldb >> 4, ks = K >> 5;
  int t = blockIdx.x * blockDim.x + threadIdx.x;
  if (t >= ntiles * ks * 32) return;
  int lane = t & 31;
  int kstep = (t >> 5) % ks;
  int nt = t / (32 * ks);
  int n = lane & 15;
  int kb = (lane & 16) ? 8 : 0;
  _Float16* op = P + (size_t)t * 16;
  const float* ip = W + (size_t)(kstep * 32 + kb) * ldb + nt * 16 + n;
#pragma unroll
  for (int e = 0; e < 16; ++e) op[e] = (_Float16)ip[(size_t)(e + (e & 8)) * ldb];
}

__global__ void k_cvt_f16(const float* __restrict__ X, _Float16* __restrict__ Y,
                          long long n) {
  long long i = (long long)blockIdx.x * blockDim.x + threadIdx.x;
  if (i < n) Y[i] = (_Float16)X[i];
}

// ---------------------------------------------------------------------------
// GEMM1: H1[N,256] = X[N,128] @ W1[128,256]
// block=256 (8 waves), 16 rows; wave -> two 16-col tiles. Packed W1 in LDS.
// ---------------------------------------------------------------------------
__global__ void k_gemm1(const _Float16* __restrict__ X, const _Float16* __restrict__ WP,
                        float* __restrict__ H, int N) {
  extern __shared__ char smem[];
  stage_lds(WP, smem, 16 * 4 * 32 * 32, 256);       // 64KB packed W1
  int wave = threadIdx.x >> 5, lane = threadIdx.x & 31;
  int row0 = blockIdx.x * 16;
  int nt0 = wave * 2;
  const v16h* BW = (const v16h*)smem;
  v8f acc0 = {}; v8f acc1 = {};
#pragma unroll
  for (int ks = 0; ks < 4; ++ks) {
    v16h a  = load_a_frag_h(X, 128, row0, ks * 32, lane, N);
    v16h b0 = BW[(nt0 * 4 + ks) * 32 + lane];
    v16h b1 = BW[((nt0 + 1) * 4 + ks) * 32 + lane];
    acc0 = __builtin_amdgcn_wmma_f32_16x16x32_f16(false, a, false, b0, (short)0, acc0, false, false);
    acc1 = __builtin_amdgcn_wmma_f32_16x16x32_f16(false, a, false, b1, (short)0, acc1, false, false);
  }
  store_c_frag(H, 256, row0, nt0 * 16,      lane, acc0, N);
  store_c_frag(H, 256, row0, nt0 * 16 + 16, lane, acc1, N);
}

// ---------------------------------------------------------------------------
// GEMM2: H2[N,64] = R[N,256] @ W2[256,64]
// block=128 (4 waves), 16 rows; wave -> one 16-col tile. Packed W2 in LDS.
// ---------------------------------------------------------------------------
__global__ void k_gemm2(const _Float16* __restrict__ R, const _Float16* __restrict__ WP,
                        float* __restrict__ H, int N) {
  extern __shared__ char smem[];
  stage_lds(WP, smem, 4 * 8 * 32 * 32, 128);        // 32KB packed W2
  int wave = threadIdx.x >> 5, lane = threadIdx.x & 31;
  int row0 = blockIdx.x * 16;
  const v16h* BW = (const v16h*)smem;
  v8f acc = {};
#pragma unroll
  for (int ks = 0; ks < 8; ++ks) {
    v16h a = load_a_frag_h(R, 256, row0, ks * 32, lane, N);
    v16h b = BW[(wave * 8 + ks) * 32 + lane];
    acc = __builtin_amdgcn_wmma_f32_16x16x32_f16(false, a, false, b, (short)0, acc, false, false);
  }
  store_c_frag(H, 64, row0, wave * 16, lane, acc, N);
}

// ---------------------------------------------------------------------------
// Utility kernels
// ---------------------------------------------------------------------------
__global__ void k_fill_f32(float* __restrict__ p, float v, int n) {
  int i = blockIdx.x * blockDim.x + threadIdx.x;
  if (i < n) p[i] = v;
}

__global__ void k_reduce_sum(const float* __restrict__ x, int n, float* __restrict__ acc) {
  __shared__ float sdata[256];
  float s = 0.f;
  for (int i = blockIdx.x * blockDim.x + threadIdx.x; i < n; i += gridDim.x * blockDim.x)
    s += x[i];
  sdata[threadIdx.x] = s;
  __syncthreads();
  for (int st = 128; st > 0; st >>= 1) {
    if ((int)threadIdx.x < st) sdata[threadIdx.x] += sdata[threadIdx.x + st];
    __syncthreads();
  }
  if (threadIdx.x == 0) atomicAdd(acc, sdata[0]);
}

// scal layout: [0]=sum_ea, [1]=mean_ea, [2..5]=ce1[h], [6]=ce2
__global__ void k_consts(const float* __restrict__ We1, const float* __restrict__ ae1,
                         const float* __restrict__ We2, const float* __restrict__ ae2,
                         float* __restrict__ scal, int E) {
  int t = threadIdx.x;
  if (t < 4) {
    float s = 0.f;
    for (int c = 0; c < 64; ++c) s += We1[t * 64 + c] * ae1[t * 64 + c];
    scal[2 + t] = s;
  } else if (t == 4) {
    float s = 0.f;
    for (int c = 0; c < 64; ++c) s += We2[c] * ae2[c];
    scal[6] = s;
  } else if (t == 5) {
    scal[1] = scal[0] / (float)E;
  }
}

// ---------------------------------------------------------------------------
// Node attention logits
// ---------------------------------------------------------------------------
__global__ void k_node_logits1(const float* __restrict__ H1, const float* __restrict__ as1,
                               const float* __restrict__ ad1, float* __restrict__ al_s,
                               float* __restrict__ al_d, int N) {
  int t = blockIdx.x * blockDim.x + threadIdx.x;
  if (t >= N * 4) return;
  int n = t >> 2, h = t & 3;
  const float* hp = H1 + (size_t)n * 256 + h * 64;
  float s = 0.f, d = 0.f;
#pragma unroll 4
  for (int c = 0; c < 64; ++c) {
    float v = hp[c];
    s += v * as1[h * 64 + c];
    d += v * ad1[h * 64 + c];
  }
  al_s[t] = s; al_d[t] = d;
}

__global__ void k_node_logits2(const float* __restrict__ H2, const float* __restrict__ as2,
                               const float* __restrict__ ad2, float* __restrict__ al_s,
                               float* __restrict__ al_d, int N) {
  int n = blockIdx.x * blockDim.x + threadIdx.x;
  if (n >= N) return;
  const float* hp = H2 + (size_t)n * 64;
  float s = 0.f, d = 0.f;
#pragma unroll 4
  for (int c = 0; c < 64; ++c) {
    float v = hp[c];
    s += v * as2[c];
    d += v * ad2[c];
  }
  al_s[n] = s; al_d[n] = d;
}

// ---------------------------------------------------------------------------
// Edge kernels
// ---------------------------------------------------------------------------
__device__ __forceinline__ void edge_sd(const int* __restrict__ ei,
                                        const float* __restrict__ ea,
                                        const float* __restrict__ scal,
                                        int e, int E, int& s, int& d, float& a) {
  if (e < E) { s = ei[e]; d = ei[E + e]; a = ea[e]; }
  else       { s = d = e - E; a = scal[1]; }
}

__device__ __forceinline__ void atomicMaxF(float* addr, float val) {
  if (val >= 0.f) atomicMax((int*)addr, __float_as_int(val));
  else            atomicMin((unsigned int*)addr, __float_as_uint(val));
}

__global__ void k_edge_alpha1(const int* __restrict__ ei, const float* __restrict__ ea,
                              const float* __restrict__ al_s, const float* __restrict__ al_d,
                              const float* __restrict__ scal, float* __restrict__ alpha,
                              float* __restrict__ m, int E, int EP) {
  int e = blockIdx.x * blockDim.x + threadIdx.x;
  if (e >= EP) return;
  int s, d; float eav;
  edge_sd(ei, ea, scal, e, E, s, d, eav);
#pragma unroll
  for (int h = 0; h < 4; ++h) {
    float al = al_s[s * 4 + h] + al_d[d * 4 + h] + eav * scal[2 + h];
    al = (al > 0.f) ? al : NEG_SLOPE * al;
    alpha[(size_t)e * 4 + h] = al;
    atomicMaxF(&m[d * 4 + h], al);
  }
}

__global__ void k_edge_exp1(const int* __restrict__ ei, const float* __restrict__ ea,
                            const float* __restrict__ scal, float* __restrict__ alpha,
                            const float* __restrict__ m, float* __restrict__ den,
                            int E, int EP) {
  int e = blockIdx.x * blockDim.x + threadIdx.x;
  if (e >= EP) return;
  int s, d; float eav;
  edge_sd(ei, ea, scal, e, E, s, d, eav);
#pragma unroll
  for (int h = 0; h < 4; ++h) {
    float ex = __expf(alpha[(size_t)e * 4 + h] - m[d * 4 + h]);
    alpha[(size_t)e * 4 + h] = ex;
    atomicAdd(&den[d * 4 + h], ex);
  }
}

// one thread = (edge, 4 channels); 64 threads per edge cover 256 channels
__global__ void k_edge_aggr1(const int* __restrict__ ei, const float* __restrict__ ea,
                             const float* __restrict__ scal, const float* __restrict__ alpha,
                             const float* __restrict__ den, const float* __restrict__ H1,
                             float* __restrict__ out, int E, int EP) {
  long long t = (long long)blockIdx.x * blockDim.x + threadIdx.x;
  if (t >= (long long)EP * 64) return;
  int e = (int)(t >> 6);
  int c0 = ((int)t & 63) << 2;
  int h = c0 >> 6;
  int s, d; float eav;
  edge_sd(ei, ea, scal, e, E, s, d, eav);
  float a = alpha[(size_t)e * 4 + h] / (den[d * 4 + h] + 1e-16f);
  const float4 hv = *(const float4*)(H1 + (size_t)s * 256 + c0);
  float* op = out + (size_t)d * 256 + c0;
  atomicAdd(op + 0, hv.x * a);
  atomicAdd(op + 1, hv.y * a);
  atomicAdd(op + 2, hv.z * a);
  atomicAdd(op + 3, hv.w * a);
}

// bias + relu, writing f16 GEMM2 input directly
__global__ void k_bias_relu_h(const float* __restrict__ out, const float* __restrict__ b,
                              _Float16* __restrict__ rh, long long total) {
  long long t = (long long)blockIdx.x * blockDim.x + threadIdx.x;
  if (t >= total) return;
  float v = out[t] + b[(int)t & 255];
  rh[t] = (_Float16)(v > 0.f ? v : 0.f);
}

__global__ void k_edge_alpha2(const int* __restrict__ ei, const float* __restrict__ ea,
                              const float* __restrict__ al_s, const float* __restrict__ al_d,
                              const float* __restrict__ scal, float* __restrict__ alpha,
                              float* __restrict__ m, int E, int EP) {
  int e = blockIdx.x * blockDim.x + threadIdx.x;
  if (e >= EP) return;
  int s, d; float eav;
  edge_sd(ei, ea, scal, e, E, s, d, eav);
  float al = al_s[s] + al_d[d] + eav * scal[6];
  al = (al > 0.f) ? al : NEG_SLOPE * al;
  alpha[e] = al;
  atomicMaxF(&m[d], al);
}

__global__ void k_edge_exp2(const int* __restrict__ ei, const float* __restrict__ ea,
                            const float* __restrict__ scal, float* __restrict__ alpha,
                            const float* __restrict__ m, float* __restrict__ den,
                            int E, int EP) {
  int e = blockIdx.x * blockDim.x + threadIdx.x;
  if (e >= EP) return;
  int s, d; float eav;
  edge_sd(ei, ea, scal, e, E, s, d, eav);
  float ex = __expf(alpha[e] - m[d]);
  alpha[e] = ex;
  atomicAdd(&den[d], ex);
}

// one thread = (edge, 4 channels); 16 threads per edge cover 64 channels
__global__ void k_edge_aggr2(const int* __restrict__ ei, const float* __restrict__ ea,
                             const float* __restrict__ scal, const float* __restrict__ alpha,
                             const float* __restrict__ den, const float* __restrict__ H2,
                             float* __restrict__ out, int E, int EP) {
  long long t = (long long)blockIdx.x * blockDim.x + threadIdx.x;
  if (t >= (long long)EP * 16) return;
  int e = (int)(t >> 4);
  int c0 = ((int)t & 15) << 2;
  int s, d; float eav;
  edge_sd(ei, ea, scal, e, E, s, d, eav);
  float a = alpha[e] / (den[d] + 1e-16f);
  const float4 hv = *(const float4*)(H2 + (size_t)s * 64 + c0);
  float* op = out + (size_t)d * 64 + c0;
  atomicAdd(op + 0, hv.x * a);
  atomicAdd(op + 1, hv.y * a);
  atomicAdd(op + 2, hv.z * a);
  atomicAdd(op + 3, hv.w * a);
}

// ---------------------------------------------------------------------------
// bias + LayerNorm + mean-pool accumulate (one wave per node, lane = 2 chans)
// ---------------------------------------------------------------------------
__global__ void k_ln_pool(const float* __restrict__ out2, const float* __restrict__ b2,
                          const float* __restrict__ ln_g, const float* __restrict__ ln_b,
                          const int* __restrict__ batch, float* __restrict__ sums,
                          float* __restrict__ cnt, int N) {
  int wave = threadIdx.x >> 5;
  int lane = threadIdx.x & 31;
  int node = blockIdx.x * 8 + wave;
  if (node >= N) return;
  float v0 = out2[(size_t)node * 64 + lane]      + b2[lane];
  float v1 = out2[(size_t)node * 64 + lane + 32] + b2[lane + 32];
  float s = v0 + v1;
#pragma unroll
  for (int mk = 16; mk > 0; mk >>= 1) s += __shfl_xor(s, mk, 32);
  float mu = s * (1.f / 64.f);
  float d0 = v0 - mu, d1 = v1 - mu;
  float vs = d0 * d0 + d1 * d1;
#pragma unroll
  for (int mk = 16; mk > 0; mk >>= 1) vs += __shfl_xor(vs, mk, 32);
  float inv = rsqrtf(vs * (1.f / 64.f) + 1e-5f);
  float h0 = d0 * inv * ln_g[lane]      + ln_b[lane];
  float h1 = d1 * inv * ln_g[lane + 32] + ln_b[lane + 32];
  int g = batch[node];
  atomicAdd(&sums[g * 64 + lane],      h0);
  atomicAdd(&sums[g * 64 + lane + 32], h1);
  if (lane == 0) atomicAdd(&cnt[g], 1.0f);
}

// ---------------------------------------------------------------------------
// Fusion MLP head: one thread per graph (G=64)
// d_out: risk[0..G) | graph_embedding[G .. G+G*64) | latent[G+G*64 .. +G*32)
// ---------------------------------------------------------------------------
__global__ void k_fusion(const float* __restrict__ sums, const float* __restrict__ cnt,
                         const float* __restrict__ clin, const float* __restrict__ met,
                         const float* __restrict__ Wf1, const float* __restrict__ bf1,
                         const float* __restrict__ Wf2, const float* __restrict__ bf2,
                         const float* __restrict__ Wr, const float* __restrict__ br,
                         float* __restrict__ dout, int G) {
  int g = blockIdx.x * blockDim.x + threadIdx.x;
  if (g >= G) return;
  float inv = 1.f / fmaxf(cnt[g], 1.0f);
  float fused[224];
  for (int c = 0; c < 64; ++c) {
    float e = sums[g * 64 + c] * inv;
    fused[c] = e;
    dout[G + g * 64 + c] = e;                      // graph_embedding
  }
  for (int c = 0; c < 32; ++c)  fused[64 + c] = clin[g * 32 + c];
  for (int c = 0; c < 128; ++c) fused[96 + c] = met[g * 128 + c];
  float lat1[64];
  for (int j = 0; j < 64; ++j) {
    float s = bf1[j];
    for (int k = 0; k < 224; ++k) s += fused[k] * Wf1[k * 64 + j];
    lat1[j] = s > 0.f ? s : 0.f;
  }
  float risk = br[0];
  for (int j = 0; j < 32; ++j) {
    float s = bf2[j];
    for (int k = 0; k < 64; ++k) s += lat1[k] * Wf2[k * 32 + j];
    s = s > 0.f ? s : 0.f;
    dout[G + G * 64 + g * 32 + j] = s;             // latent
    risk += s * Wr[j];
  }
  dout[g] = risk;                                  // risk
}

// ---------------------------------------------------------------------------
// Host launcher
// ---------------------------------------------------------------------------
extern "C" void kernel_launch(void* const* d_in, const int* in_sizes, int n_in,
                              void* d_out, int out_size, void* d_ws, size_t ws_size,
                              hipStream_t stream) {
  const float* x     = (const float*)d_in[0];
  const int*   ei    = (const int*)  d_in[1];
  const float* eattr = (const float*)d_in[2];
  const int*   batch = (const int*)  d_in[3];
  const float* clin  = (const float*)d_in[4];
  const float* met   = (const float*)d_in[5];
  const float* W1    = (const float*)d_in[6];
  const float* as1   = (const float*)d_in[7];
  const float* ad1   = (const float*)d_in[8];
  const float* ae1   = (const float*)d_in[9];
  const float* We1   = (const float*)d_in[10];
  const float* b1    = (const float*)d_in[11];
  const float* W2    = (const float*)d_in[12];
  const float* as2   = (const float*)d_in[13];
  const float* ad2   = (const float*)d_in[14];
  const float* ae2   = (const float*)d_in[15];
  const float* We2   = (const float*)d_in[16];
  const float* b2    = (const float*)d_in[17];
  const float* ln_g  = (const float*)d_in[18];
  const float* ln_b  = (const float*)d_in[19];
  const float* Wf1   = (const float*)d_in[20];
  const float* bf1   = (const float*)d_in[21];
  const float* Wf2   = (const float*)d_in[22];
  const float* bf2   = (const float*)d_in[23];
  const float* Wr    = (const float*)d_in[24];
  const float* br    = (const float*)d_in[25];

  const int N  = in_sizes[0] / 128;
  const int E  = in_sizes[2];
  const int G  = in_sizes[4] / 32;
  const int EP = E + N;

  // ---- workspace layout (float slots) ----
  float* ws    = (float*)d_ws;
  float* bufA  = ws;                                  // N*256 : h1; later h2pre(N*64)+out2(N*64)
  float* bufB  = bufA + (size_t)N * 256;              // N*256 : out1 (f32 accumulators)
  float* alphaB= bufB + (size_t)N * 256;              // EP*4
  float* al_s1 = alphaB + (size_t)EP * 4;             // N*4
  float* al_d1 = al_s1 + (size_t)N * 4;
  float* m1    = al_d1 + (size_t)N * 4;
  float* den1  = m1    + (size_t)N * 4;
  float* al_s2 = den1  + (size_t)N * 4;               // N
  float* al_d2 = al_s2 + N;
  float* m2    = al_d2 + N;
  float* den2  = m2 + N;
  float* sums  = den2 + N;                            // G*64
  float* cnt   = sums + (size_t)G * 64;               // G
  float* scal  = cnt + G;                             // 8
  // f16 region: xh (N*128 halfs) and rh (N*256 halfs) have disjoint lifetimes
  _Float16* hbuf = (_Float16*)(scal + 8);             // N*256 halfs (shared)
  _Float16* xh   = hbuf;                              // N*128 halfs (gemm1 input)
  _Float16* rh   = hbuf;                              // N*256 halfs (gemm2 input)
  _Float16* w1p  = hbuf + (size_t)N * 256;            // 16*4*32*16 = 32768 halfs
  _Float16* w2p  = w1p + 32768;                       // 4*8*32*16  = 16384 halfs

  float* h2pre = bufA;                                // N*64 (reuses h1 space)
  float* out2  = bufA + (size_t)N * 64;               // N*64

  const float NEG_BIG = -3.0e38f;

  // zero-init accumulators that are live from the start
  (void)hipMemsetAsync(bufB, 0, (size_t)N * 256 * sizeof(float), stream);     // out1
  (void)hipMemsetAsync(den1, 0, (size_t)N * 4 * sizeof(float), stream);
  (void)hipMemsetAsync(den2, 0, (size_t)N * sizeof(float), stream);
  (void)hipMemsetAsync(sums, 0, ((size_t)G * 64 + G) * sizeof(float), stream); // sums+cnt
  (void)hipMemsetAsync(scal, 0, 8 * sizeof(float), stream);
  k_fill_f32<<<(N * 4 + 255) / 256, 256, 0, stream>>>(m1, NEG_BIG, N * 4);
  k_fill_f32<<<(N + 255) / 256, 256, 0, stream>>>(m2, NEG_BIG, N);

  // edge_attr mean + attention-edge scalar constants
  k_reduce_sum<<<512, 256, 0, stream>>>(eattr, E, scal);
  k_consts<<<1, 64, 0, stream>>>(We1, ae1, We2, ae2, scal, E);

  // pre-pack weights to WMMA fragment order (f16) and convert x to f16
  k_pack_w<<<(16 * 4 * 32 + 255) / 256, 256, 0, stream>>>(W1, w1p, 128, 256);
  k_pack_w<<<(4 * 8 * 32 + 255) / 256, 256, 0, stream>>>(W2, w2p, 256, 64);
  {
    long long tot = (long long)N * 128;
    k_cvt_f16<<<(unsigned)((tot + 255) / 256), 256, 0, stream>>>(x, xh, tot);
  }

  // ---- Layer 1 ----
  k_gemm1<<<(N + 15) / 16, 256, 65536, stream>>>(xh, w1p, bufA, N);
  k_node_logits1<<<(N * 4 + 255) / 256, 256, 0, stream>>>(bufA, as1, ad1, al_s1, al_d1, N);
  k_edge_alpha1<<<(EP + 255) / 256, 256, 0, stream>>>(ei, eattr, al_s1, al_d1, scal, alphaB, m1, E, EP);
  k_edge_exp1<<<(EP + 255) / 256, 256, 0, stream>>>(ei, eattr, scal, alphaB, m1, den1, E, EP);
  {
    long long tot = (long long)EP * 64;
    k_edge_aggr1<<<(unsigned)((tot + 255) / 256), 256, 0, stream>>>(ei, eattr, scal, alphaB, den1, bufA, bufB, E, EP);
  }
  {
    long long tot = (long long)N * 256;
    k_bias_relu_h<<<(unsigned)((tot + 255) / 256), 256, 0, stream>>>(bufB, b1, rh, tot);
  }

  // ---- Layer 2 (h2pre/out2 reuse bufA; h1 is dead after aggr1) ----
  k_gemm2<<<(N + 15) / 16, 128, 32768, stream>>>(rh, w2p, h2pre, N);
  (void)hipMemsetAsync(out2, 0, (size_t)N * 64 * sizeof(float), stream);
  k_node_logits2<<<(N + 255) / 256, 256, 0, stream>>>(h2pre, as2, ad2, al_s2, al_d2, N);
  k_edge_alpha2<<<(EP + 255) / 256, 256, 0, stream>>>(ei, eattr, al_s2, al_d2, scal, alphaB, m2, E, EP);
  k_edge_exp2<<<(EP + 255) / 256, 256, 0, stream>>>(ei, eattr, scal, alphaB, m2, den2, E, EP);
  {
    long long tot = (long long)EP * 16;
    k_edge_aggr2<<<(unsigned)((tot + 255) / 256), 256, 0, stream>>>(ei, eattr, scal, alphaB, den2, h2pre, out2, E, EP);
  }

  // ---- LayerNorm + mean pool ----
  k_ln_pool<<<(N + 7) / 8, 256, 0, stream>>>(out2, b2, ln_g, ln_b, batch, sums, cnt, N);

  // ---- fusion MLP head ----
  k_fusion<<<(G + 63) / 64, 64, 0, stream>>>(sums, cnt, clin, met, Wf1, bf1, Wf2, bf2,
                                             Wr, br, (float*)d_out, G);
}
